// MSHGTModel_42339787604850
// MI455X (gfx1250) — matrix operand order repaired and checked by reading, not compiled, not run on hardware
//
#include <hip/hip_runtime.h>
#include <hip/hip_bf16.h>
#include <math.h>

// ---------------------------------------------------------------------------
// MSHGT model forward for MI455X (gfx1250, wave32, WMMA).
// Dense contractions -> v_wmma_f32_16x16x32_f16 with double-buffered LDS
// staging (f32 global -> b128 loads -> packed cvt -> f16 LDS). Attention is
// flash-style; pe-loss GEMM fused with sigmoid+MSE (S never materialized).
// Degree-bias tile staged global->LDS via GLOBAL_LOAD_ASYNC_TO_LDS_B128.
// ---------------------------------------------------------------------------

static constexpr int kNA    = 4096;
static constexpr int kNPA   = 8192;
static constexpr int kNT    = 4096;
static constexpr int kHeads = 8;
static constexpr int kD     = 128;

typedef __attribute__((ext_vector_type(16))) _Float16 v16h;
typedef __attribute__((ext_vector_type(8)))  _Float16 v8h;
typedef __attribute__((ext_vector_type(8)))  float    v8f;

// ====================== staging helpers (device) ===========================
// Stage a 64x32 f32->f16 tile from row-major P[R x K] (rows base..base+63,
// cols k0..k0+31). fast: no bounds checks, float4 loads, 16B LDS store.
__device__ __forceinline__ void stage_rowmajor(
    const float* __restrict__ P, _Float16 (*T)[32],
    int base, int k0, int R, int K, int tid, bool fast)
{
    int row = tid >> 2;
    int kc  = (tid & 3) * 8;
    int gr  = base + row;
    if (fast) {
        const float* p = P + (size_t)gr * K + k0 + kc;
        __builtin_prefetch(p + 32, 0, 1);          // -> global_prefetch_b8
        float4 f0 = *(const float4*)p;
        float4 f1 = *(const float4*)(p + 4);
        v8h h;
        h[0] = (_Float16)f0.x; h[1] = (_Float16)f0.y;
        h[2] = (_Float16)f0.z; h[3] = (_Float16)f0.w;
        h[4] = (_Float16)f1.x; h[5] = (_Float16)f1.y;
        h[6] = (_Float16)f1.z; h[7] = (_Float16)f1.w;
        *(v8h*)&T[row][kc] = h;
    } else {
#pragma unroll 1
        for (int j = 0; j < 8; ++j) {
            int gk = k0 + kc + j;
            T[row][kc + j] = (gr < R && gk < K) ? (_Float16)P[(size_t)gr * K + gk]
                                                : (_Float16)0.f;
        }
    }
}

// Stage B[K x N] transposed into T[n][k] (cols bn..bn+63, rows k0..k0+31).
__device__ __forceinline__ void stage_colmajor(
    const float* __restrict__ B, _Float16 (*T)[32],
    int bn, int k0, int N, int K, int tid, bool fast)
{
    int kk = tid & 31;
    int nb = (tid >> 5) * 8;
    int gk = k0 + kk;
    if (fast) {
        const float* p = B + (size_t)gk * N + bn + nb;
        float4 f0 = *(const float4*)p;
        float4 f1 = *(const float4*)(p + 4);
        _Float16 h[8] = { (_Float16)f0.x, (_Float16)f0.y, (_Float16)f0.z, (_Float16)f0.w,
                          (_Float16)f1.x, (_Float16)f1.y, (_Float16)f1.z, (_Float16)f1.w };
#pragma unroll
        for (int j = 0; j < 8; ++j) T[nb + j][kk] = h[j];
    } else {
#pragma unroll 1
        for (int j = 0; j < 8; ++j) {
            int gn = bn + nb + j;
            T[nb + j][kk] = (gn < N && gk < K) ? (_Float16)B[(size_t)gk * N + gn]
                                               : (_Float16)0.f;
        }
    }
}

// WMMA fragment builders (ISA 7.12.2 layouts), tiles are [row][32] f16 in LDS.
__device__ __forceinline__ v16h frag_a(const _Float16 (*T)[32], int tr, int lr, int hi) {
    v16h a;
#pragma unroll
    for (int i = 0; i < 8; ++i) a[i]     = T[tr * 16 + lr][hi * 8 + i];
#pragma unroll
    for (int i = 0; i < 8; ++i) a[8 + i] = T[tr * 16 + lr][16 + hi * 8 + i];
    return a;
}
__device__ __forceinline__ v16h frag_b(const _Float16 (*T)[32], int tc, int lr, int hi) {
    v16h b;
#pragma unroll
    for (int i = 0; i < 8; ++i) b[i]     = T[tc * 16 + lr][hi * 8 + i];
#pragma unroll
    for (int i = 0; i < 8; ++i) b[8 + i] = T[tc * 16 + lr][16 + hi * 8 + i];
    return b;
}

// ============================ WMMA GEMM ====================================
// C[M,N] = act( A[M,K] @ B + bias[N] + addm[M,N] )
// bt==0: B is [K,N] row-major.  bt==1: B is [N,K] row-major (C = A@B^T).
__global__ __launch_bounds__(256) void wmma_gemm_kernel(
    const float* __restrict__ A, const float* __restrict__ B,
    const float* __restrict__ bias, const float* __restrict__ addm,
    float* __restrict__ C, int M, int N, int K, int bt, int relu)
{
#ifdef __gfx1250__
    __shared__ __align__(16) _Float16 As[2][64][32];
    __shared__ __align__(16) _Float16 Bs[2][64][32];
    const int tid  = threadIdx.x;
    const int lane = tid & 31;
    const int wave = tid >> 5;
    const int lr   = lane & 15;
    const int hi   = lane >> 4;
    const int bm   = blockIdx.y * 64;
    const int bn   = blockIdx.x * 64;
    const int t0   = wave * 2;
    const int tr   = t0 >> 2;
    const int tc0  = t0 & 3;
    const int tc1  = tc0 + 1;
    v8f acc0 = {}; v8f acc1 = {};

    const bool fA  = ((K & 3) == 0) && (bm + 64 <= M);
    const bool fBT = ((K & 3) == 0) && (bn + 64 <= N);
    const bool fBN = ((N & 3) == 0) && (bn + 64 <= N);
    const int  nch = (K + 31) >> 5;

    // stage chunk 0
    {
        bool kf = 32 <= K;
        stage_rowmajor(A, As[0], bm, 0, M, K, tid, fA && kf);
        if (bt) stage_rowmajor(B, Bs[0], bn, 0, N, K, tid, fBT && kf);
        else    stage_colmajor(B, Bs[0], bn, 0, N, K, tid, fBN && kf);
    }
    __syncthreads();

    int cur = 0;
    for (int it = 0; it < nch; ++it) {
        int k0n = (it + 1) << 5;
        if (it + 1 < nch) {       // prefetch next chunk into other buffer
            bool kf = (k0n + 32) <= K;
            stage_rowmajor(A, As[cur ^ 1], bm, k0n, M, K, tid, fA && kf);
            if (bt) stage_rowmajor(B, Bs[cur ^ 1], bn, k0n, N, K, tid, fBT && kf);
            else    stage_colmajor(B, Bs[cur ^ 1], bn, k0n, N, K, tid, fBN && kf);
        }
        v16h af  = frag_a(As[cur], tr, lr, hi);
        v16h bf0 = frag_b(Bs[cur], tc0, lr, hi);
        v16h bf1 = frag_b(Bs[cur], tc1, lr, hi);
        acc0 = __builtin_amdgcn_wmma_f32_16x16x32_f16(false, af, false, bf0, (short)0, acc0, false, false);
        acc1 = __builtin_amdgcn_wmma_f32_16x16x32_f16(false, af, false, bf1, (short)0, acc1, false, false);
        __syncthreads();
        cur ^= 1;
    }
    // --- epilogue ---
#pragma unroll
    for (int r = 0; r < 8; ++r) {
        int m = bm + tr * 16 + r + 8 * hi;
        if (m >= M) continue;
        int n0 = bn + tc0 * 16 + lr;
        int n1 = bn + tc1 * 16 + lr;
        if (n0 < N) {
            float v = acc0[r];
            if (bias) v += bias[n0];
            if (addm) v += addm[(size_t)m * N + n0];
            if (relu) v = fmaxf(v, 0.f);
            C[(size_t)m * N + n0] = v;
        }
        if (n1 < N) {
            float v = acc1[r];
            if (bias) v += bias[n1];
            if (addm) v += addm[(size_t)m * N + n1];
            if (relu) v = fmaxf(v, 0.f);
            C[(size_t)m * N + n1] = v;
        }
    }
#endif
}

// =================== fused  sigmoid(QK^T) MSE loss =========================
// All dims multiples of tile size by construction (4096, K=128) -> fast path.
__global__ __launch_bounds__(256) void pe_loss_kernel(
    const float* __restrict__ Qm, const float* __restrict__ Kmat,
    const float* __restrict__ Ao, float* __restrict__ loss,
    int Nn, int Kd, float scale)
{
#ifdef __gfx1250__
    __shared__ __align__(16) _Float16 As[2][64][32];
    __shared__ __align__(16) _Float16 Bs[2][64][32];
    const int tid = threadIdx.x, lane = tid & 31, wave = tid >> 5;
    const int lr = lane & 15, hi = lane >> 4;
    const int bm = blockIdx.y * 64, bn = blockIdx.x * 64;
    const int t0 = wave * 2, tr = t0 >> 2, tc0 = t0 & 3, tc1 = tc0 + 1;
    v8f acc0 = {}; v8f acc1 = {};
    const int nch = Kd >> 5;

    stage_rowmajor(Qm,   As[0], bm, 0, Nn, Kd, tid, true);
    stage_rowmajor(Kmat, Bs[0], bn, 0, Nn, Kd, tid, true);
    __syncthreads();
    int cur = 0;
    for (int it = 0; it < nch; ++it) {
        int k0n = (it + 1) << 5;
        if (it + 1 < nch) {
            stage_rowmajor(Qm,   As[cur ^ 1], bm, k0n, Nn, Kd, tid, true);
            stage_rowmajor(Kmat, Bs[cur ^ 1], bn, k0n, Nn, Kd, tid, true);
        }
        v16h af  = frag_a(As[cur], tr, lr, hi);
        v16h bf0 = frag_b(Bs[cur], tc0, lr, hi);
        v16h bf1 = frag_b(Bs[cur], tc1, lr, hi);
        acc0 = __builtin_amdgcn_wmma_f32_16x16x32_f16(false, af, false, bf0, (short)0, acc0, false, false);
        acc1 = __builtin_amdgcn_wmma_f32_16x16x32_f16(false, af, false, bf1, (short)0, acc1, false, false);
        __syncthreads();
        cur ^= 1;
    }
    float part = 0.f;
#pragma unroll
    for (int r = 0; r < 8; ++r) {
        int m = bm + tr * 16 + r + 8 * hi;
        int nn[2] = { bn + tc0 * 16 + lr, bn + tc1 * 16 + lr };
        float vv[2] = { acc0[r], acc1[r] };
#pragma unroll
        for (int q = 0; q < 2; ++q) {
            float s = 1.f / (1.f + __expf(-vv[q]));
#pragma unroll
            for (int i = 0; i < 4; ++i) {
                float d = s - Ao[((size_t)i * Nn + m) * Nn + nn[q]];
                part += d * d;
            }
        }
    }
#pragma unroll
    for (int mroff = 16; mroff >= 1; mroff >>= 1) part += __shfl_xor(part, mroff);
    if (lane == 0) atomicAdd(loss, part * scale);
#endif
}

// ======================= flash attention (dh=16) ===========================
__global__ __launch_bounds__(256) void attn_kernel(
    const float* __restrict__ Qf, const float* __restrict__ Kf,
    const float* __restrict__ Vf, const float* __restrict__ dbv,
    float* __restrict__ O, int N, int D)
{
#ifdef __gfx1250__
    const int h  = blockIdx.y;
    const int n0 = blockIdx.x * 64;
    __shared__ __align__(16) _Float16 Qs[64][32];
    __shared__ __align__(16) _Float16 Ks[64][32];
    __shared__ _Float16 Vt[16][64];
    __shared__ float    Ss[64][65];        // padded: avoid bank conflicts
    __shared__ _Float16 Ps[64][64];
    __shared__ float    Os[64][16];
    __shared__ __align__(16) float dbt[64];   // degree-bias tile (async-staged)
    __shared__ float    mrun[64], lrun[64];
    const int tid = threadIdx.x, lane = tid & 31, wave = tid >> 5;
    const int lr = lane & 15, hi = lane >> 4;
    const int rq = tid >> 2;               // 0..63
    const int cq = (tid & 3) * 8;          // 0,8,16,24
    const int rv = tid >> 2;               // 0..63
    const int dv = (tid & 3) * 4;          // 0,4,8,12

    {   // Q tile: scale 1/sqrt(16), pad K dim 16->32 with zeros
        if (cq < 16) {
            const float* qp = Qf + (size_t)(n0 + rq) * D + h * 16 + cq;
            float4 f0 = *(const float4*)qp;
            float4 f1 = *(const float4*)(qp + 4);
            v8h hv;
            hv[0] = (_Float16)(0.25f * f0.x); hv[1] = (_Float16)(0.25f * f0.y);
            hv[2] = (_Float16)(0.25f * f0.z); hv[3] = (_Float16)(0.25f * f0.w);
            hv[4] = (_Float16)(0.25f * f1.x); hv[5] = (_Float16)(0.25f * f1.y);
            hv[6] = (_Float16)(0.25f * f1.z); hv[7] = (_Float16)(0.25f * f1.w);
            *(v8h*)&Qs[rq][cq] = hv;
        } else {
            v8h z = {};
            *(v8h*)&Qs[rq][cq] = z;
        }
    }
    if (tid < 64) { mrun[tid] = -3.0e38f; lrun[tid] = 0.f; }
    { Os[rv][dv] = 0.f; Os[rv][dv + 1] = 0.f; Os[rv][dv + 2] = 0.f; Os[rv][dv + 3] = 0.f; }
    __syncthreads();

    for (int m0 = 0; m0 < N; m0 += 64) {
        {   // K tile
            if (cq < 16) {
                const float* kp = Kf + (size_t)(m0 + rq) * D + h * 16 + cq;
                float4 f0 = *(const float4*)kp;
                float4 f1 = *(const float4*)(kp + 4);
                v8h hv;
                hv[0] = (_Float16)f0.x; hv[1] = (_Float16)f0.y;
                hv[2] = (_Float16)f0.z; hv[3] = (_Float16)f0.w;
                hv[4] = (_Float16)f1.x; hv[5] = (_Float16)f1.y;
                hv[6] = (_Float16)f1.z; hv[7] = (_Float16)f1.w;
                *(v8h*)&Ks[rq][cq] = hv;
            } else {
                v8h z = {};
                *(v8h*)&Ks[rq][cq] = z;
            }
        }
        {   // V tile transposed [d][m]
            const float* vp = Vf + (size_t)(m0 + rv) * D + h * 16 + dv;
            float4 f = *(const float4*)vp;
            Vt[dv + 0][rv] = (_Float16)f.x;
            Vt[dv + 1][rv] = (_Float16)f.y;
            Vt[dv + 2][rv] = (_Float16)f.z;
            Vt[dv + 3][rv] = (_Float16)f.w;
        }
        // degree-bias tile: 64 f32 = 256 B, one async B128 issue on 16 lanes
        // (CDNA5 GLOBAL_LOAD_ASYNC_TO_LDS_B128, tracked by ASYNCcnt).
        if (wave == 0) {
            if (lane < 16) {
                // low 32 bits of a shared-memory generic pointer == LDS offset
                unsigned ldsoff = (unsigned)(size_t)(&dbt[0]) + (unsigned)lane * 16u;
                const float* gp = dbv + m0 + lane * 4;
                asm volatile("global_load_async_to_lds_b128 %0, %1, off"
                             :: "v"(ldsoff), "v"(gp) : "memory");
            }
            asm volatile("s_wait_asynccnt 0x0" ::: "memory");
        }
        __syncthreads();
        {   // scores S = Q K^T : 8 waves x 2 subtiles
            int t0 = wave * 2, tr = t0 >> 2, tc0 = t0 & 3, tc1 = tc0 + 1;
            v16h af  = frag_a(Qs, tr, lr, hi);
            v16h bf0 = frag_b(Ks, tc0, lr, hi);
            v16h bf1 = frag_b(Ks, tc1, lr, hi);
            v8f a0 = {}; v8f a1 = {};
            a0 = __builtin_amdgcn_wmma_f32_16x16x32_f16(false, af, false, bf0, (short)0, a0, false, false);
            a1 = __builtin_amdgcn_wmma_f32_16x16x32_f16(false, af, false, bf1, (short)0, a1, false, false);
#pragma unroll
            for (int r = 0; r < 8; ++r) {
                Ss[tr * 16 + r + 8 * hi][tc0 * 16 + lr] = a0[r];
                Ss[tr * 16 + r + 8 * hi][tc1 * 16 + lr] = a1[r];
            }
        }
        __syncthreads();
        {   // online softmax update: 4 consecutive lanes per row (same wave)
            int row = tid >> 2; int c0 = (tid & 3) * 16;
            float sv[16]; float mx = -3.0e38f;
#pragma unroll
            for (int j = 0; j < 16; ++j) {
                float s = Ss[row][c0 + j] + dbt[c0 + j];
                sv[j] = s; mx = fmaxf(mx, s);
            }
            mx = fmaxf(mx, __shfl_xor(mx, 1));
            mx = fmaxf(mx, __shfl_xor(mx, 2));
            float mold = mrun[row];
            float mnew = fmaxf(mold, mx);
            float alpha = __expf(mold - mnew);
            float ls = 0.f;
#pragma unroll
            for (int j = 0; j < 16; ++j) {
                float p = __expf(sv[j] - mnew);
                Ps[row][c0 + j] = (_Float16)p;
                ls += p;
            }
            ls += __shfl_xor(ls, 1);
            ls += __shfl_xor(ls, 2);
#pragma unroll
            for (int j = 0; j < 4; ++j) Os[row][(tid & 3) * 4 + j] *= alpha;
            if ((tid & 3) == 0) { lrun[row] = lrun[row] * alpha + ls; mrun[row] = mnew; }
        }
        __syncthreads();
        if (wave < 4) {   // O += P @ V : 4 subtiles of 16x16, K=64 in 2 chunks
            int tr = wave;
            v8f acc;
#pragma unroll
            for (int r = 0; r < 8; ++r) acc[r] = Os[tr * 16 + r + 8 * hi][lr];
#pragma unroll
            for (int kc = 0; kc < 2; ++kc) {
                v16h af;
#pragma unroll
                for (int i = 0; i < 8; ++i) af[i]     = Ps[tr * 16 + lr][kc * 32 + hi * 8 + i];
#pragma unroll
                for (int i = 0; i < 8; ++i) af[8 + i] = Ps[tr * 16 + lr][kc * 32 + 16 + hi * 8 + i];
                v16h bf;
#pragma unroll
                for (int i = 0; i < 8; ++i) bf[i]     = Vt[lr][kc * 32 + hi * 8 + i];
#pragma unroll
                for (int i = 0; i < 8; ++i) bf[8 + i] = Vt[lr][kc * 32 + 16 + hi * 8 + i];
                acc = __builtin_amdgcn_wmma_f32_16x16x32_f16(false, af, false, bf, (short)0, acc, false, false);
            }
#pragma unroll
            for (int r = 0; r < 8; ++r) Os[tr * 16 + r + 8 * hi][lr] = acc[r];
        }
        __syncthreads();
    }
    {   // write O row-block, float4 stores
        float l = lrun[rv];
        float inv = 1.f / (l > 0.f ? l : 1.f);
        float4 o4;
        o4.x = Os[rv][dv + 0] * inv;
        o4.y = Os[rv][dv + 1] * inv;
        o4.z = Os[rv][dv + 2] * inv;
        o4.w = Os[rv][dv + 3] * inv;
        *(float4*)(O + (size_t)(n0 + rv) * D + h * 16 + dv) = o4;
    }
#endif
}

// ====================== small / elementwise kernels ========================
__global__ void k_fill_f32(float* p, float v, int n) {
    int i = blockIdx.x * 256 + threadIdx.x; if (i < n) p[i] = v;
}
__global__ void k_fill_u32(unsigned* p, unsigned v, int n) {
    int i = blockIdx.x * 256 + threadIdx.x; if (i < n) p[i] = v;
}
// C must be a power of two
__global__ void k_bias_init(float* o, const float* b1, const float* b2, int n, int C) {
    int i = blockIdx.x * 256 + threadIdx.x;
    if (i < n) o[i] = b1[i & (C - 1)] + (b2 ? b2[i & (C - 1)] : 0.f);
}
__global__ void k_relu(float* p, int n) {
    int i = blockIdx.x * 256 + threadIdx.x; if (i < n) p[i] = fmaxf(p[i], 0.f);
}
__global__ void k_db(const int* deg, float* dbv, int n) {
    int i = blockIdx.x * 256 + threadIdx.x;
    if (i < n) dbv[i] = logf(fmaxf((float)deg[i], 1.f));
}
__global__ void k_concat3(const float* a, const float* b, const float* c,
                          float* o, int N, int D) {
    int i = blockIdx.x * 256 + threadIdx.x;
    if (i >= N * 3 * D) return;
    int n = i / (3 * D); int j = i % (3 * D);
    const float* s = (j < D) ? a : ((j < 2 * D) ? b : c);
    o[i] = s[(size_t)n * D + (j % D)];
}
__global__ void k_softmax4(const float* lg, float* o, int N) {
    int n = blockIdx.x * 256 + threadIdx.x;
    if (n >= N) return;
    const float* r = lg + 4 * n;
    float mx = fmaxf(fmaxf(r[0], r[1]), fmaxf(r[2], r[3]));
    float e0 = __expf(r[0] - mx), e1 = __expf(r[1] - mx);
    float e2 = __expf(r[2] - mx), e3 = __expf(r[3] - mx);
    float s = e0 + e1 + e2 + e3;
    o[4 * n + 0] = e0 / s; o[4 * n + 1] = e1 / s;
    o[4 * n + 2] = e2 / s; o[4 * n + 3] = e3 / s;
}

// wave-per-(n,h): out[n,h] = sum_c Hm[n, h*C + c] * a[h*C + c]; H = 1<<hshift
__global__ __launch_bounds__(256) void scores_kernel(
    const float* __restrict__ Hm, const float* __restrict__ a,
    float* __restrict__ out, int N, int hshift, int C)
{
    int wave = threadIdx.x >> 5, lane = threadIdx.x & 31;
    int idx = blockIdx.x * 8 + wave;
    if (idx >= (N << hshift)) return;
    int n = idx >> hshift, h = idx & ((1 << hshift) - 1);
    float s = 0.f;
    for (int c = lane; c < C; c += 32)
        s += Hm[(size_t)n * (C << hshift) + h * C + c] * a[h * C + c];
#pragma unroll
    for (int m = 16; m >= 1; m >>= 1) s += __shfl_xor(s, m);
    if (lane == 0) out[idx] = s;
}

// order-preserving float->uint encoding for atomic max
__device__ inline unsigned fenc(float f) {
    unsigned u = __float_as_uint(f);
    return (u & 0x80000000u) ? ~u : (u | 0x80000000u);
}
__device__ inline float fdec(unsigned u) {
    u = (u & 0x80000000u) ? (u & 0x7fffffffu) : ~u;
    return __uint_as_float(u);
}
__device__ inline float lrelu02(float x) { return x > 0.f ? x : 0.2f * x; }

__global__ void edge_max_kernel(const int* src, const int* dst, const float* ss,
                                const float* sd, unsigned* menc, int E, int hshift) {
    long long t = (long long)blockIdx.x * 256 + threadIdx.x;
    if (t >= ((long long)E << hshift)) return;
    int e = (int)(t >> hshift), h = (int)(t & ((1 << hshift) - 1));
    int H = 1 << hshift;
    float ev = lrelu02(ss[src[e] * H + h] + sd[dst[e] * H + h]);
    atomicMax(&menc[dst[e] * H + h], fenc(ev));
}
__global__ void mfix_kernel(const unsigned* menc, float* m, int n) {
    int i = blockIdx.x * 256 + threadIdx.x;
    if (i >= n) return;
    float v = fdec(menc[i]);
    m[i] = isinf(v) ? 0.f : v;
}
__global__ void edge_den_kernel(const int* src, const int* dst, const float* ss,
                                const float* sd, const float* m, float* den,
                                int E, int hshift) {
    long long t = (long long)blockIdx.x * 256 + threadIdx.x;
    if (t >= ((long long)E << hshift)) return;
    int e = (int)(t >> hshift), h = (int)(t & ((1 << hshift) - 1));
    int H = 1 << hshift;
    float ev = lrelu02(ss[src[e] * H + h] + sd[dst[e] * H + h]);
    atomicAdd(&den[dst[e] * H + h], __expf(ev - m[dst[e] * H + h]));
}
// one thread per (edge, head, channel); C = 1<<cshift, H = 1<<hshift
__global__ void edge_scatter_kernel(const int* src, const int* dst, const float* ss,
                                    const float* sd, const float* m, const float* den,
                                    const float* Hs, float* out, int E, int hshift,
                                    int cshift) {
    long long t = (long long)blockIdx.x * 256 + threadIdx.x;
    if (t >= ((long long)E << (hshift + cshift))) return;
    int C = 1 << cshift, H = 1 << hshift;
    int c = (int)(t & (C - 1));
    long long eh = t >> cshift;
    int h = (int)(eh & (H - 1));
    int e = (int)(eh >> hshift);
    int s = src[e], d = dst[e];
    float ev = lrelu02(ss[s * H + h] + sd[d * H + h]);
    float alpha = __expf(ev - m[d * H + h]) / (den[d * H + h] + 1e-16f);
    atomicAdd(&out[((size_t)d << (hshift + cshift)) + h * C + c],
              Hs[((size_t)s << (hshift + cshift)) + h * C + c] * alpha);
}

// y = LN(x + r) * g + b   (wave per row, D=128)
__global__ __launch_bounds__(256) void ln_kernel(
    const float* __restrict__ x, const float* __restrict__ r,
    const float* __restrict__ g, const float* __restrict__ b,
    float* __restrict__ y, int N, int D)
{
    int wave = threadIdx.x >> 5, lane = threadIdx.x & 31;
    int row = blockIdx.x * 8 + wave;
    if (row >= N) return;
    float v[4]; float s = 0.f;
#pragma unroll
    for (int j = 0; j < 4; ++j) {
        int d = lane + j * 32;
        float t = x[(size_t)row * D + d] + (r ? r[(size_t)row * D + d] : 0.f);
        v[j] = t; s += t;
    }
#pragma unroll
    for (int m = 16; m >= 1; m >>= 1) s += __shfl_xor(s, m);
    float mean = s / D;
    float vs = 0.f;
#pragma unroll
    for (int j = 0; j < 4; ++j) { float d0 = v[j] - mean; vs += d0 * d0; }
#pragma unroll
    for (int m = 16; m >= 1; m >>= 1) vs += __shfl_xor(vs, m);
    float inv = rsqrtf(vs / D + 1e-5f);
#pragma unroll
    for (int j = 0; j < 4; ++j) {
        int d = lane + j * 32;
        y[(size_t)row * D + d] = (v[j] - mean) * inv * g[d] + b[d];
    }
}

// ============================ host driver ==================================
static inline void launch_gemm(const float* A, const float* B, const float* bias,
                               const float* addm, float* C, int M, int N, int K,
                               int bt, int relu, hipStream_t s) {
    dim3 g((N + 63) / 64, (M + 63) / 64);
    wmma_gemm_kernel<<<g, dim3(256), 0, s>>>(A, B, bias, addm, C, M, N, K, bt, relu);
}

static inline void run_edges(const int* src, const int* dst, const float* ss,
                             const float* sd, unsigned* menc, float* mval, float* den,
                             const float* Hs, float* out, int E, int hshift,
                             int n_dst, hipStream_t s) {
    int nm = n_dst << hshift;
    k_fill_u32<<<(nm + 255) / 256, 256, 0, s>>>(menc, 0x007fffffu /* enc(-inf) */, nm);
    k_fill_f32<<<(nm + 255) / 256, 256, 0, s>>>(den, 0.f, nm);
    long long t1 = (long long)E << hshift;
    edge_max_kernel<<<(unsigned)((t1 + 255) / 256), 256, 0, s>>>(src, dst, ss, sd, menc, E, hshift);
    mfix_kernel<<<(nm + 255) / 256, 256, 0, s>>>(menc, mval, nm);
    edge_den_kernel<<<(unsigned)((t1 + 255) / 256), 256, 0, s>>>(src, dst, ss, sd, mval, den, E, hshift);
    long long t2 = (long long)E << (hshift + 7);
    edge_scatter_kernel<<<(unsigned)((t2 + 255) / 256), 256, 0, s>>>(
        src, dst, ss, sd, mval, den, Hs, out, E, hshift, 7);
}

extern "C" void kernel_launch(void* const* d_in, const int* in_sizes, int n_in,
                              void* d_out, int out_size, void* d_ws, size_t ws_size,
                              hipStream_t stream) {
    // ---- inputs (setup_inputs dict order) ----
    const float* x_author = (const float*)d_in[0];
    const float* x_paper  = (const float*)d_in[1];
    const float* x_term   = (const float*)d_in[2];
    const int* ap_src = (const int*)d_in[3];  const int* ap_dst = (const int*)d_in[4];
    const int* pa_src = (const int*)d_in[5];  const int* pa_dst = (const int*)d_in[6];
    const int* tp_src = (const int*)d_in[9];  const int* tp_dst = (const int*)d_in[10];
    const float* origA = (const float*)d_in[11];
    const int*   deg   = (const int*)d_in[12];
    const float* t1_W = (const float*)d_in[13]; const float* t1_b = (const float*)d_in[14];
    const float* t2_W = (const float*)d_in[15]; const float* t2_b = (const float*)d_in[16];
    const float* t3_W = (const float*)d_in[17]; const float* t3_b = (const float*)d_in[18];
    const float* g1W  = (const float*)d_in[19]; // [4,128,1024]
    const float* g1as = (const float*)d_in[20]; // [4,8,128]
    const float* g1ad = (const float*)d_in[21];
    const float* g1b  = (const float*)d_in[22]; // [4,1024]
    const float* g2W  = (const float*)d_in[23]; // [4,1024,128]
    const float* g2as = (const float*)d_in[24]; // [4,1,128]
    const float* g2ad = (const float*)d_in[25];
    const float* g2b  = (const float*)d_in[26]; // [4,128]
    const float* pe_embed = (const float*)d_in[27];
    const float* peWQ = (const float*)d_in[28];
    const float* peWK = (const float*)d_in[29];
    const float* gtWq = (const float*)d_in[30]; const float* gtbq = (const float*)d_in[31];
    const float* gtWk = (const float*)d_in[32]; const float* gtbk = (const float*)d_in[33];
    const float* gtWv = (const float*)d_in[34]; const float* gtbv = (const float*)d_in[35];
    const float* gtWo = (const float*)d_in[36]; const float* gtbo = (const float*)d_in[37];
    const float* gtg1 = (const float*)d_in[38]; const float* gtbe1 = (const float*)d_in[39];
    const float* gtWf1 = (const float*)d_in[40]; const float* gtbf1 = (const float*)d_in[41];
    const float* gtWf2 = (const float*)d_in[42]; const float* gtbf2 = (const float*)d_in[43];
    const float* gtg2 = (const float*)d_in[44]; const float* gtbe2 = (const float*)d_in[45];
    const float* catW = (const float*)d_in[46]; const float* catb = (const float*)d_in[47];
    const float* mW1 = (const float*)d_in[48]; const float* mb1 = (const float*)d_in[49];
    const float* mW2 = (const float*)d_in[50]; const float* mb2 = (const float*)d_in[51];

    const int E_ap = in_sizes[3], E_pa = in_sizes[5], E_tp = in_sizes[9];

    // ---- workspace bump allocator ----
    size_t off = 0;
    auto alloc = [&](size_t nfloats) -> float* {
        float* p = (float*)((char*)d_ws + off);
        off += ((nfloats * sizeof(float) + 255) / 256) * 256;
        return p;
    };
    float* xa     = alloc((size_t)kNA * kD);
    float* xp     = alloc((size_t)kNPA * kD);
    float* xt     = alloc((size_t)kNT * kD);
    float* bigA   = alloc((size_t)kNPA * 1024);   // Hd / Hs staging
    float* bigB   = alloc((size_t)kNPA * 1024);
    float* acc_p1 = alloc((size_t)kNPA * 1024);
    float* acc_a1 = alloc((size_t)kNA * 1024);
    float* ssb    = alloc((size_t)kNPA * kHeads);
    float* sdb    = alloc((size_t)kNPA * kHeads);
    unsigned* menc = (unsigned*)alloc((size_t)kNPA * kHeads);
    float* mval   = alloc((size_t)kNPA * kHeads);
    float* den    = alloc((size_t)kNPA * kHeads);
    float* Hs2    = alloc((size_t)kNPA * kD);
    float* Hd2    = alloc((size_t)kNA * kD);
    float* outs0  = alloc((size_t)kNA * kD);
    float* outs1  = alloc((size_t)kNA * kD);
    float* outs2  = alloc((size_t)kNA * kD);
    float* peQ    = alloc((size_t)kNA * kD);
    float* peK    = alloc((size_t)kNA * kD);
    float* dbv    = alloc((size_t)kNA);
    float* Qb     = alloc((size_t)kNA * kD);
    float* Kb     = alloc((size_t)kNA * kD);
    float* Vb     = alloc((size_t)kNA * kD);
    float* Ob     = alloc((size_t)kNA * kD);
    float* x1buf  = alloc((size_t)kNA * kD);
    float* tmpD   = alloc((size_t)kNA * kD);
    float* tmp256 = alloc((size_t)kNA * 256);
    float* catbuf = alloc((size_t)kNA * 384);
    float* logits = alloc((size_t)kNA * 4);
    if (off > ws_size) return;  // workspace too small: do nothing (deterministic)

    float* out_f = (float*)d_out;
    float* probs = out_f + 1;
    float* xgt   = out_f + 1 + (size_t)kNA * 4;

    // ---- 1. feature transforms (ReLU(x@W+b)) ----
    launch_gemm(x_author, t1_W, t1_b, nullptr, xa, kNA, kD, 334, 0, 1, stream);
    launch_gemm(x_paper,  t2_W, t2_b, nullptr, xp, kNPA, kD, 512, 0, 1, stream);
    launch_gemm(x_term,   t3_W, t3_b, nullptr, xt, kNT, kD, 128, 0, 1, stream);

    // ---- 2. GAT layer 1 into papers: (a->p type0) + (t->p type3) ----
    {
        int n = kNPA * 1024;
        k_bias_init<<<(n + 255) / 256, 256, 0, stream>>>(acc_p1, g1b + 0 * 1024, g1b + 3 * 1024, n, 1024);
    }
    // type 0: author -> paper
    launch_gemm(xp, g1W + (size_t)0 * 128 * 1024, nullptr, nullptr, bigA, kNPA, 1024, 128, 0, 0, stream);
    scores_kernel<<<(kNPA * kHeads + 7) / 8, 256, 0, stream>>>(bigA, g1ad + 0 * 8 * 128, sdb, kNPA, 3, 128);
    launch_gemm(xa, g1W + (size_t)0 * 128 * 1024, nullptr, nullptr, bigB, kNA, 1024, 128, 0, 0, stream);
    scores_kernel<<<(kNA * kHeads + 7) / 8, 256, 0, stream>>>(bigB, g1as + 0 * 8 * 128, ssb, kNA, 3, 128);
    run_edges(ap_src, ap_dst, ssb, sdb, menc, mval, den, bigB, acc_p1, E_ap, 3, kNPA, stream);
    // type 3: term -> paper
    launch_gemm(xp, g1W + (size_t)3 * 128 * 1024, nullptr, nullptr, bigA, kNPA, 1024, 128, 0, 0, stream);
    scores_kernel<<<(kNPA * kHeads + 7) / 8, 256, 0, stream>>>(bigA, g1ad + 3 * 8 * 128, sdb, kNPA, 3, 128);
    launch_gemm(xt, g1W + (size_t)3 * 128 * 1024, nullptr, nullptr, bigB, kNT, 1024, 128, 0, 0, stream);
    scores_kernel<<<(kNT * kHeads + 7) / 8, 256, 0, stream>>>(bigB, g1as + 3 * 8 * 128, ssb, kNT, 3, 128);
    run_edges(tp_src, tp_dst, ssb, sdb, menc, mval, den, bigB, acc_p1, E_tp, 3, kNPA, stream);
    // type 1: paper -> author
    {
        int n = kNA * 1024;
        k_bias_init<<<(n + 255) / 256, 256, 0, stream>>>(acc_a1, g1b + 1 * 1024, nullptr, n, 1024);
    }
    launch_gemm(xa, g1W + (size_t)1 * 128 * 1024, nullptr, nullptr, bigB, kNA, 1024, 128, 0, 0, stream);
    scores_kernel<<<(kNA * kHeads + 7) / 8, 256, 0, stream>>>(bigB, g1ad + 1 * 8 * 128, sdb, kNA, 3, 128);
    launch_gemm(xp, g1W + (size_t)1 * 128 * 1024, nullptr, nullptr, bigA, kNPA, 1024, 128, 0, 0, stream);
    scores_kernel<<<(kNPA * kHeads + 7) / 8, 256, 0, stream>>>(bigA, g1as + 1 * 8 * 128, ssb, kNPA, 3, 128);
    run_edges(pa_src, pa_dst, ssb, sdb, menc, mval, den, bigA, acc_a1, E_pa, 3, kNA, stream);
    // ReLU both
    k_relu<<<(kNPA * 1024 + 255) / 256, 256, 0, stream>>>(acc_p1, kNPA * 1024);
    k_relu<<<(kNA * 1024 + 255) / 256, 256, 0, stream>>>(acc_a1, kNA * 1024);

    // ---- 3. GAT layer 2 (paper -> author, heads=1, outc=128) -> outs0 ----
    {
        int n = kNA * kD;
        k_bias_init<<<(n + 255) / 256, 256, 0, stream>>>(outs0, g2b + 1 * 128, nullptr, n, 128);
    }
    launch_gemm(acc_a1, g2W + (size_t)1 * 1024 * 128, nullptr, nullptr, Hd2, kNA, 128, 1024, 0, 0, stream);
    scores_kernel<<<(kNA + 7) / 8, 256, 0, stream>>>(Hd2, g2ad + 1 * 128, sdb, kNA, 0, 128);
    launch_gemm(acc_p1, g2W + (size_t)1 * 1024 * 128, nullptr, nullptr, Hs2, kNPA, 128, 1024, 0, 0, stream);
    scores_kernel<<<(kNPA + 7) / 8, 256, 0, stream>>>(Hs2, g2as + 1 * 128, ssb, kNPA, 0, 128);
    run_edges(pa_src, pa_dst, ssb, sdb, menc, mval, den, Hs2, outs0, E_pa, 0, kNA, stream);

    // ---- 4. positional encodings + fused adjacency loss ----
    launch_gemm(pe_embed, peWQ, nullptr, nullptr, peQ, kNA, kD, 128, 0, 0, stream);
    launch_gemm(pe_embed, peWK, nullptr, nullptr, peK, kNA, kD, 128, 0, 0, stream);
    k_fill_f32<<<1, 256, 0, stream>>>(out_f, 0.f, 1);
    {
        dim3 g(kNA / 64, kNA / 64);
        float scale = 1.f / (4.f * (float)kNA * (float)kNA);
        pe_loss_kernel<<<g, dim3(256), 0, stream>>>(peQ, peK, origA, out_f, kNA, 128, scale);
    }
    k_db<<<(kNA + 255) / 256, 256, 0, stream>>>(deg, dbv, kNA);

    // ---- 5. graph transformer layers ----
    float* cur = outs0;
    float* outn[2] = { outs1, outs2 };
    for (int i = 0; i < 2; ++i) {
        const size_t wOff = (size_t)i * 128 * 128;
        launch_gemm(cur, gtWq + wOff, gtbq + i * 128, peQ, Qb, kNA, kD, 128, 0, 0, stream);
        launch_gemm(cur, gtWk + wOff, gtbk + i * 128, peK, Kb, kNA, kD, 128, 0, 0, stream);
        launch_gemm(cur, gtWv + wOff, gtbv + i * 128, nullptr, Vb, kNA, kD, 128, 0, 0, stream);
        attn_kernel<<<dim3(kNA / 64, kHeads), dim3(256), 0, stream>>>(Qb, Kb, Vb, dbv, Ob, kNA, kD);
        launch_gemm(Ob, gtWo + wOff, gtbo + i * 128, nullptr, tmpD, kNA, kD, 128, 0, 0, stream);
        ln_kernel<<<(kNA + 7) / 8, 256, 0, stream>>>(cur, tmpD, gtg1 + i * 128, gtbe1 + i * 128, x1buf, kNA, kD);
        launch_gemm(x1buf, gtWf1 + (size_t)i * 128 * 256, gtbf1 + i * 256, nullptr, tmp256, kNA, 256, 128, 0, 1, stream);
        launch_gemm(tmp256, gtWf2 + (size_t)i * 256 * 128, gtbf2 + i * 128, nullptr, tmpD, kNA, kD, 256, 0, 0, stream);
        ln_kernel<<<(kNA + 7) / 8, 256, 0, stream>>>(x1buf, tmpD, gtg2 + i * 128, gtbe2 + i * 128, outn[i], kNA, kD);
        cur = outn[i];
    }

    // ---- 6. concat head + MLP + softmax ----
    k_concat3<<<(kNA * 384 + 255) / 256, 256, 0, stream>>>(outs0, outs1, outs2, catbuf, kNA, kD);
    launch_gemm(catbuf, catW, catb, nullptr, xgt, kNA, kD, 384, 0, 1, stream);
    launch_gemm(xgt, mW1, mb1, nullptr, tmp256, kNA, 256, 128, 0, 1, stream);
    launch_gemm(tmp256, mW2, mb2, nullptr, logits, kNA, 4, 256, 0, 0, stream);
    k_softmax4<<<(kNA + 255) / 256, 256, 0, stream>>>(logits, probs, kNA);
}
